// RQVAE_87711822119408
// MI455X (gfx1250) — matrix operand order repaired and checked by reading, not compiled, not run on hardware
//
#include <hip/hip_runtime.h>
#include <hip/hip_bf16.h>
#include <cstdint>
#include <cstddef>

// Problem sizes (match reference)
#define NROWS   16384
#define IN_DIM  768
#define CDIM    256
#define KCODES  4096
#define TILE_PITCH 264   // 256 bf16 + 8 pad (16B) to break LDS bank alignment

typedef __bf16 bf16;
typedef __attribute__((ext_vector_type(16))) __bf16 v16bf;
typedef __attribute__((ext_vector_type(8)))  __bf16 v8bf;
typedef __attribute__((ext_vector_type(4)))  __bf16 v4bf;
typedef __attribute__((ext_vector_type(8)))  float  v8f;

static __device__ __forceinline__ v16bf cat16(v8bf lo, v8bf hi) {
  return __builtin_shufflevector(lo, hi, 0,1,2,3,4,5,6,7,8,9,10,11,12,13,14,15);
}

// ---------------------------------------------------------------------------
// Prep kernels
// ---------------------------------------------------------------------------
__global__ void cvt_bf16_kernel(const float* __restrict__ src,
                                bf16* __restrict__ dst, size_t n) {
  size_t i = (size_t)blockIdx.x * blockDim.x + threadIdx.x;
  size_t stride = (size_t)gridDim.x * blockDim.x;
  for (; i < n; i += stride) dst[i] = (bf16)src[i];
}

// src is R x C (row-major f32); dst is C x R (row-major bf16) == src^T
__global__ void transpose_bf16_kernel(const float* __restrict__ src,
                                      bf16* __restrict__ dst, int R, int C) {
  size_t n = (size_t)R * C;
  size_t i = (size_t)blockIdx.x * blockDim.x + threadIdx.x;
  size_t stride = (size_t)gridDim.x * blockDim.x;
  for (; i < n; i += stride) {
    int r = (int)(i / C), c = (int)(i % C);
    dst[(size_t)c * R + r] = (bf16)src[i];
  }
}

__global__ void enorm_kernel(const float* __restrict__ E,
                             float* __restrict__ nrm, int K, int C) {
  int k = blockIdx.x * blockDim.x + threadIdx.x;
  if (k >= K) return;
  float s = 0.f;
  const float* row = E + (size_t)k * C;
  for (int c = 0; c < C; ++c) { float e = row[c]; s += e * e; }
  nrm[k] = s;
}

// ---------------------------------------------------------------------------
// Generic bf16 WMMA GEMM: out = A(MxK) * Bt^T + bias, Bt is N x K row-major.
// One 16x16 D tile per wave; f32 output + optional bf16 copy.
// ---------------------------------------------------------------------------
__global__ __launch_bounds__(256) void gemm_bias_wmma(
    const bf16* __restrict__ A, const bf16* __restrict__ Bt,
    const float* __restrict__ bias, float* __restrict__ outF,
    bf16* __restrict__ outH, int M, int Ncols, int K) {
  int tid  = threadIdx.x;
  int lane = tid & 31;
  int g    = lane >> 4;      // half-wave select (ISA fragment layout)
  int lr   = lane & 15;      // row (A) / col (B,C,D) within tile
  int wave = (int)((blockIdx.x * (size_t)blockDim.x + tid) >> 5);
  int tilesN = Ncols >> 4;
  int mTile = wave / tilesN;
  int nTile = wave % tilesN;
  if (mTile * 16 >= M) return;

  const bf16* arow = A  + (size_t)(mTile * 16 + lr) * K;
  const bf16* brow = Bt + (size_t)(nTile * 16 + lr) * K + g * 16;

  v8f acc = {};
  int kSteps = K >> 5;
  for (int kf = 0; kf < kSteps; ++kf) {
    int k0 = kf << 5;
    // A 16x32 bf16 fragment: lane holds row lr, contiguous K [g*16,+16)
    v8bf lo = *(const v8bf*)(arow + k0 + g * 16);
    v8bf hi = *(const v8bf*)(arow + k0 + g * 16 + 8);
    v16bf a = cat16(lo, hi);
    // B 32x16 bf16 fragment: lane holds col lr, contiguous K [g*16,+16)
    v16bf b = *(const v16bf*)(brow + k0);
    acc = __builtin_amdgcn_wmma_f32_16x16x32_bf16(false, a, false, b,
                                                  (short)0, acc, false, false);
  }

  float bv = bias ? bias[nTile * 16 + lr] : 0.f;
#pragma unroll
  for (int v = 0; v < 8; ++v) {
    int r = mTile * 16 + g * 8 + v;   // D layout: VGPR v -> M = v + g*8
    int c = nTile * 16 + lr;
    float val = acc[v] + bv;
    size_t o = (size_t)r * Ncols + c;
    outF[o] = val;
    if (outH) outH[o] = (bf16)val;
  }
}

// ---------------------------------------------------------------------------
// Fused VQ stage: distances via WMMA on async-LDS-staged E tiles, argmin,
// gather, residual, zq_sum accumulation.
// Block = 256 threads = 8 waves; each wave owns 16 rows; block owns 128 rows.
// E tiles (16 codes x 256 K bf16 = 8KB) are double-buffered in LDS via
// GLOBAL_LOAD_ASYNC_TO_LDS_B128 (ASYNCcnt), overlapping fetch with WMMA.
// ---------------------------------------------------------------------------
__global__ __launch_bounds__(256) void vq_stage_wmma(
    const float* __restrict__ zeF, const bf16* __restrict__ zeH,
    const float* __restrict__ Ef,  const bf16* __restrict__ Eh,
    const float* __restrict__ eNorm,
    int* __restrict__ nnOut, float* __restrict__ zqOut,
    float* __restrict__ zeNextF, bf16* __restrict__ zeNextH,
    float* __restrict__ zqSumF, bf16* __restrict__ zqSumH, int stage) {
  __shared__ __align__(16) bf16 s_tile[2][16 * TILE_PITCH];
  __shared__ int s_idx[128];

  int tid    = threadIdx.x;
  int lane   = tid & 31;
  int g      = lane >> 4;
  int lr     = lane & 15;
  int waveId = tid >> 5;
  int rowBase = blockIdx.x * 128 + waveId * 16;

  // Preload A fragments for this wave's 16 rows, full K=256 (8 x 16x16x32).
  const bf16* zr = zeH + (size_t)(rowBase + lr) * CDIM;
  v16bf afrag[8];
#pragma unroll
  for (int kf = 0; kf < 8; ++kf) {
    int k0 = kf << 5;
    v8bf lo = *(const v8bf*)(zr + k0 + g * 16);
    v8bf hi = *(const v8bf*)(zr + k0 + g * 16 + 8);
    afrag[kf] = cat16(lo, hi);
  }

  // Async-stage one 16-code tile of E into LDS buffer `buf`.
  // 512 chunks of 16B; 256 threads x 2 chunks; 2 async ops per wave per tile.
  auto issue_tile = [&](int nt, int buf) {
    const bf16* gbase = Eh + (size_t)nt * 16 * CDIM;
#pragma unroll
    for (int h = 0; h < 2; ++h) {
      int c = tid + h * 256;
      int r = c >> 5;          // code row within tile (32 chunks per row)
      int inner = c & 31;      // 16B chunk within row
      const bf16* gp = gbase + r * CDIM + inner * 8;
      unsigned ldsOff =
          (unsigned)(uintptr_t)(&s_tile[buf][r * TILE_PITCH + inner * 8]);
      asm volatile("global_load_async_to_lds_b128 %0, %1, off"
                   :: "v"(ldsOff), "v"((unsigned long long)(uintptr_t)gp)
                   : "memory");
    }
  };

  float bestV[8];
  int   bestI[8];
#pragma unroll
  for (int v = 0; v < 8; ++v) { bestV[v] = 3.4e38f; bestI[v] = 0; }

  issue_tile(0, 0);

  // Sweep all codes in tiles of 16; ||ze||^2 is row-constant -> dropped.
  for (int nt = 0; nt < KCODES / 16; ++nt) {
    int cur = nt & 1;
    if (nt + 1 < KCODES / 16) {
      issue_tile(nt + 1, cur ^ 1);
      // Retire previous tile's 2 ops; next tile's 2 may remain in flight.
      asm volatile("s_wait_asynccnt 0x2" ::: "memory");
    } else {
      asm volatile("s_wait_asynccnt 0x0" ::: "memory");
    }
    __syncthreads();   // tile `nt` visible in LDS to all waves

    int code = nt * 16 + lr;
    const bf16* trow = &s_tile[cur][lr * TILE_PITCH + g * 16];
    v8f acc = {};
#pragma unroll
    for (int kf = 0; kf < 8; ++kf) {
      acc = __builtin_amdgcn_wmma_f32_16x16x32_bf16(
          false, afrag[kf], false, *(const v16bf*)(trow + (kf << 5)),
          (short)0, acc, false, false);
    }
    float en = eNorm[code];
#pragma unroll
    for (int v = 0; v < 8; ++v) {
      float d = en - 2.0f * acc[v];         // argmin-equivalent distance
      if (d < bestV[v]) { bestV[v] = d; bestI[v] = code; }
    }
    __syncthreads();   // all waves done reading buf[cur] before it is reused
  }

  // Butterfly argmin across the 16 lanes that share a row group.
  for (int off = 8; off > 0; off >>= 1) {
#pragma unroll
    for (int v = 0; v < 8; ++v) {
      float ov = __shfl_xor(bestV[v], off, 32);
      int   oi = __shfl_xor(bestI[v], off, 32);
      if (ov < bestV[v] || (ov == bestV[v] && oi < bestI[v])) {
        bestV[v] = ov; bestI[v] = oi;
      }
    }
  }

  if (lr == 0) {
#pragma unroll
    for (int v = 0; v < 8; ++v) {
      int r = waveId * 16 + g * 8 + v;
      s_idx[r] = bestI[v];
      nnOut[blockIdx.x * 128 + r] = bestI[v];
    }
  }
  __syncthreads();

  // Gather zq = E[idx] (f32), residual, zq_sum accumulation.
  for (int rr = 0; rr < 16; ++rr) {
    int lrow = waveId * 16 + rr;
    int grow = blockIdx.x * 128 + lrow;
    int idx  = s_idx[lrow];
#pragma unroll
    for (int j = 0; j < 2; ++j) {
      int c = j * 128 + lane * 4;
      size_t eo = (size_t)idx  * CDIM + c;
      size_t zo = (size_t)grow * CDIM + c;
      float4 q = *(const float4*)(Ef + eo);
      float4 z = *(const float4*)(zeF + zo);
      *(float4*)(zqOut + zo) = q;
      float4 res; res.x = z.x - q.x; res.y = z.y - q.y;
                  res.z = z.z - q.z; res.w = z.w - q.w;
      if (zeNextF) *(float4*)(zeNextF + zo) = res;
      if (zeNextH) {
        v4bf h; h[0] = (bf16)res.x; h[1] = (bf16)res.y;
                h[2] = (bf16)res.z; h[3] = (bf16)res.w;
        *(v4bf*)(zeNextH + zo) = h;
      }
      float4 s;
      if (stage == 1) { s = q; }
      else {
        float4 p = *(const float4*)(zqSumF + zo);
        s.x = p.x + q.x; s.y = p.y + q.y; s.z = p.z + q.z; s.w = p.w + q.w;
      }
      *(float4*)(zqSumF + zo) = s;
      if (zqSumH) {
        v4bf hs; hs[0] = (bf16)s.x; hs[1] = (bf16)s.y;
                 hs[2] = (bf16)s.z; hs[3] = (bf16)s.w;
        *(v4bf*)(zqSumH + zo) = hs;
      }
    }
  }
}

// ---------------------------------------------------------------------------
// Host launcher
// ---------------------------------------------------------------------------
extern "C" void kernel_launch(void* const* d_in, const int* in_sizes, int n_in,
                              void* d_out, int out_size, void* d_ws, size_t ws_size,
                              hipStream_t stream) {
  (void)in_sizes; (void)n_in; (void)out_size; (void)ws_size;
  const float* x     = (const float*)d_in[0];
  const float* W_enc = (const float*)d_in[1];
  const float* b_enc = (const float*)d_in[2];
  const float* Ecb[4] = {(const float*)d_in[3], (const float*)d_in[4],
                         (const float*)d_in[5], (const float*)d_in[6]};
  const float* W_dec = (const float*)d_in[7];
  const float* b_dec = (const float*)d_in[8];

  // d_out layout: x_hat | ze1..ze4 | zq1..zq4 | nn1..nn4
  float* out = (float*)d_out;
  float* x_hat = out;
  size_t off = (size_t)NROWS * IN_DIM;
  float* ze[4]; float* zq[4]; int* nn[4];
  for (int i = 0; i < 4; ++i) ze[i] = out + off + (size_t)i * NROWS * CDIM;
  off += (size_t)4 * NROWS * CDIM;
  for (int i = 0; i < 4; ++i) zq[i] = out + off + (size_t)i * NROWS * CDIM;
  off += (size_t)4 * NROWS * CDIM;
  for (int i = 0; i < 4; ++i) nn[i] = (int*)(out + off + (size_t)i * NROWS);

  // Workspace carve (256B aligned regions)
  char* w = (char*)d_ws;
  auto carve = [&](size_t bytes) -> char* {
    char* r = w; w += (bytes + 255) & ~(size_t)255; return r;
  };
  bf16* xH     = (bf16*)carve((size_t)NROWS * IN_DIM * 2);
  bf16* wEncT  = (bf16*)carve((size_t)CDIM * IN_DIM * 2);   // 256 x 768
  bf16* wDecT  = (bf16*)carve((size_t)IN_DIM * CDIM * 2);   // 768 x 256
  bf16* eH[4]; float* eN[4];
  for (int i = 0; i < 4; ++i) eH[i] = (bf16*)carve((size_t)KCODES * CDIM * 2);
  for (int i = 0; i < 4; ++i) eN[i] = (float*)carve((size_t)KCODES * 4);
  bf16*  zeH    = (bf16*)carve((size_t)NROWS * CDIM * 2);
  float* zqSumF = (float*)carve((size_t)NROWS * CDIM * 4);
  bf16*  zqSumH = (bf16*)carve((size_t)NROWS * CDIM * 2);

  // --- prep ---
  cvt_bf16_kernel<<<2048, 256, 0, stream>>>(x, xH, (size_t)NROWS * IN_DIM);
  transpose_bf16_kernel<<<768, 256, 0, stream>>>(W_enc, wEncT, IN_DIM, CDIM);
  transpose_bf16_kernel<<<768, 256, 0, stream>>>(W_dec, wDecT, CDIM, IN_DIM);
  for (int i = 0; i < 4; ++i) {
    cvt_bf16_kernel<<<1024, 256, 0, stream>>>(Ecb[i], eH[i], (size_t)KCODES * CDIM);
    enorm_kernel<<<KCODES / 256, 256, 0, stream>>>(Ecb[i], eN[i], KCODES, CDIM);
  }

  // --- encoder: ze1 = x @ W_enc + b_enc (f32 out + bf16 copy for VQ) ---
  gemm_bias_wmma<<<(NROWS / 16) * (CDIM / 16) / 8, 256, 0, stream>>>(
      xH, wEncT, b_enc, ze[0], zeH, NROWS, CDIM, IN_DIM);

  // --- 4 fused VQ stages ---
  for (int s = 0; s < 4; ++s) {
    vq_stage_wmma<<<NROWS / 128, 256, 0, stream>>>(
        ze[s], zeH, Ecb[s], eH[s], eN[s],
        nn[s], zq[s],
        (s < 3) ? ze[s + 1] : nullptr,          // ze_{s+2} output (f32)
        (s < 3) ? zeH : nullptr,                // next-stage bf16 residual
        zqSumF,
        (s == 3) ? zqSumH : nullptr,            // decoder input (bf16)
        s + 1);
  }

  // --- decoder: x_hat = zq_sum @ W_dec + b_dec ---
  gemm_bias_wmma<<<(NROWS / 16) * (IN_DIM / 16) / 8, 256, 0, stream>>>(
      zqSumH, wDecT, b_dec, x_hat, nullptr, NROWS, IN_DIM, CDIM);
}
